// QLSTMHybrid_65481071401495
// MI455X (gfx1250) — compile-verified, matrix operand order checked by reading, not run on hardware
//
#include <hip/hip_runtime.h>
#include <hip/hip_bf16.h>
#include <cstdint>
#include <cstddef>

// ---------------------------------------------------------------------------
// QLSTMHybrid forward for MI455X (gfx1250, wave32, WMMA bf16 path)
// T=1024, B=128, D=256, H=256. Gates packed as [f|i|u|o] along N (4H=1024).
//
// Phase A (pack):  W[512,256] f32 x4 -> Wt[1024][512] bf16 (transposed);
//                  X[T,B,D] f32 -> Xbf bf16 (bulk, bandwidth-bound).
// Phase B (xproj): gx = Xbf_chunk @ Wt[:, :256]^T + bias  (WMMA bf16;
//                  A from global b128, B from LDS-staged slice). gx is stored
//                  in RAW WMMA-TILE format: tile (rblk, cblk) = 256 f32 laid
//                  out lane-major (lane*8 + v), so producer stores and
//                  consumer loads are contiguous b128 pairs per lane.
// Phase C (scan):  per-batch-group (16 rows) sequential scan; per step
//                  acc = gx_tile + h @ Wt[:, 256:]^T (WMMA bf16, A from LDS),
//                  fused sigmoid/tanh gates, c in VGPRs, h -> LDS bf16.
// ---------------------------------------------------------------------------

typedef __attribute__((ext_vector_type(16))) __bf16 v16bf;
typedef __attribute__((ext_vector_type(8)))  float  v8f;

#define T_STEPS 1024
#define BSZ     128
#define DIM     256
#define HID     256
#define NGATE   4
#define GCOLS   (NGATE * HID)   // 1024
#define KTOT    (DIM + HID)     // 512
#define TC      128             // timesteps per chunk
#define NCHUNK  (T_STEPS / TC)
#define HSTR    264             // LDS row stride in bf16 elems (256 + 8 pad)
#define NCB     (GCOLS / 16)    // 64 column-blocks in gx tile grid

union ABFrag {
    uint4          q[2];
    v16bf          v;
    unsigned short s[16];
};

__device__ __forceinline__ unsigned short f2bf(float x) {
    union { float f; unsigned u; } v; v.f = x;
    unsigned r = v.u + 0x7FFFu + ((v.u >> 16) & 1u);   // round-to-nearest-even
    return (unsigned short)(r >> 16);
}
__device__ __forceinline__ float sigmoidf_(float x) {
    return 1.0f / (1.0f + __expf(-x));
}
__device__ __forceinline__ float tanhf_(float x) {
    float e = __expf(2.0f * x);
    return (e - 1.0f) / (e + 1.0f);
}

// --------------------------- Phase A1: weight pack --------------------------
// Wt[(g*HID + j) * KTOT + k] = bf16( Wg[k*HID + j] )   (writes coalesced in k)
__global__ void pack_weights(const float* __restrict__ Wf, const float* __restrict__ Wi,
                             const float* __restrict__ Wu, const float* __restrict__ Wo,
                             unsigned short* __restrict__ Wt) {
    int idx = blockIdx.x * blockDim.x + threadIdx.x;
    if (idx >= NGATE * KTOT * HID) return;
    int k = idx % KTOT;
    int j = (idx / KTOT) % HID;
    int g = idx / (KTOT * HID);
    const float* W = (g == 0) ? Wf : (g == 1) ? Wi : (g == 2) ? Wu : Wo;
    Wt[idx] = f2bf(W[k * HID + j]);
}

// --------------------------- Phase A2: x convert ----------------------------
// Bulk f32 -> bf16, identity layout. 8 elements per thread, b128 in / b128 out.
__global__ void __launch_bounds__(256)
xconvert(const float* __restrict__ x, unsigned short* __restrict__ xb) {
    size_t i = ((size_t)blockIdx.x * blockDim.x + threadIdx.x) * 8;
    const float4* p = reinterpret_cast<const float4*>(x + i);
    float4 f0 = p[0], f1 = p[1];
    ABFrag o;
    o.s[0] = f2bf(f0.x); o.s[1] = f2bf(f0.y); o.s[2] = f2bf(f0.z); o.s[3] = f2bf(f0.w);
    o.s[4] = f2bf(f1.x); o.s[5] = f2bf(f1.y); o.s[6] = f2bf(f1.z); o.s[7] = f2bf(f1.w);
    *reinterpret_cast<uint4*>(xb + i) = o.q[0];
}

// --------------------------- Phase B: x projection --------------------------
// grid = 1024 blocks x 512 threads (16 waves). All waves of a block share one
// N-quad (64 gate columns); its 32KB weight slice is staged in LDS once.
// Wave handles one M-block (16 rows) x 4 N-blocks; K = 256 -> 32 WMMAs.
__global__ void __launch_bounds__(512)
xproj_kernel(const unsigned short* __restrict__ xb,  // [T*B, D] bf16
             const unsigned short* __restrict__ Wt,  // [GCOLS][KTOT] bf16
             const float* __restrict__ bf_, const float* __restrict__ bi_,
             const float* __restrict__ bu_, const float* __restrict__ bo_,
             float* __restrict__ gx,                 // raw tiles [1024][64][256]
             int t0) {
    __shared__ unsigned short wsh[64 * HSTR];        // 64 local cols x 256 K (padded)
    const int lane   = threadIdx.x & 31;
    const int wave   = threadIdx.x >> 5;
    const int lane15 = lane & 15;
    const int laneHi = lane >> 4;
    const int mblk   = ((int)(blockIdx.x & 63) << 4) + wave;  // 0..1023
    const int nquad  = blockIdx.x >> 6;                       // 0..15
    const int row    = (mblk << 4) + lane15;                  // chunk-local row
    const unsigned short* xrow = xb + (size_t)(t0 * BSZ + row) * DIM;

    // Stage the x-part weight slice (cols nquad*64 .. +64, K 0..255) into LDS.
    for (int idx = threadIdx.x; idx < 64 * (DIM / 8); idx += 512) {
        int cl = idx >> 5;                 // local col 0..63
        int k8 = (idx & 31) << 3;          // K in steps of 8
        const uint4* src = reinterpret_cast<const uint4*>(
            Wt + (size_t)(nquad * 64 + cl) * KTOT + k8);
        *reinterpret_cast<uint4*>(&wsh[cl * HSTR + k8]) = src[0];
    }
    __syncthreads();

    const float* bs[4] = { bf_, bi_, bu_, bo_ };
    v8f acc[4];
#pragma unroll
    for (int nb = 0; nb < 4; ++nb) {
        int col  = ((nquad * 4 + nb) << 4) + lane15;
        float bv = bs[col >> 8][col & 255];
#pragma unroll
        for (int v = 0; v < 8; ++v) acc[nb][v] = bv;
    }

#pragma unroll
    for (int kb = 0; kb < DIM / 32; ++kb) {
        const int k0 = kb * 32 + laneHi * 8;
        const uint4* ap = reinterpret_cast<const uint4*>(xrow + k0);
        ABFrag a;
        a.q[0] = ap[0];   // k0 .. k0+7
        a.q[1] = ap[2];   // k0+16 .. k0+23
#pragma unroll
        for (int nb = 0; nb < 4; ++nb) {
            int cl = (nb << 4) + lane15;   // local col 0..63
            const uint4* wp = reinterpret_cast<const uint4*>(&wsh[cl * HSTR + k0]);
            ABFrag b;
            b.q[0] = wp[0];
            b.q[1] = wp[2];
            acc[nb] = __builtin_amdgcn_wmma_f32_16x16x32_bf16(
                false, a.v, false, b.v, (short)0, acc[nb], false, false);
        }
    }

    // Store raw WMMA tiles: tile (mblk, cblk), lane-major, b128 x2 per lane.
#pragma unroll
    for (int nb = 0; nb < 4; ++nb) {
        int cblk = nquad * 4 + nb;
        float4* d4 = reinterpret_cast<float4*>(
            gx + ((size_t)mblk * NCB + cblk) * 256 + lane * 8);
        d4[0] = make_float4(acc[nb][0], acc[nb][1], acc[nb][2], acc[nb][3]);
        d4[1] = make_float4(acc[nb][4], acc[nb][5], acc[nb][6], acc[nb][7]);
    }
}

// ----------------------------- Phase C: scan --------------------------------
// grid = 8 blocks x 512 threads. Block bg owns batch rows [16*bg, 16*bg+16).
// Wave w owns hidden columns [16*w, 16*w+16) (all 4 gates). h kept in LDS as
// bf16 (A matrix), c kept in 8 persistent VGPRs per lane.
__global__ void __launch_bounds__(512)
lstm_scan(const float* __restrict__ gx,              // raw tiles [1024][64][256]
          const unsigned short* __restrict__ Wt,     // [GCOLS][KTOT]
          float* __restrict__ out,                   // [T,B,H] ++ hT ++ cT
          unsigned short* __restrict__ h_state,      // [B, H] bf16
          float* __restrict__ c_state,               // [B, H] f32
          int t0) {
    __shared__ unsigned short hsh[16 * HSTR];
    const int lane   = threadIdx.x & 31;
    const int wave   = threadIdx.x >> 5;
    const int lane15 = lane & 15;
    const int laneHi = lane >> 4;
    const int bg     = blockIdx.x;
    const int col    = (wave << 4) + lane15;         // hidden column 0..255

    // Load (or zero) persistent state.
    for (int idx = threadIdx.x; idx < 16 * HID; idx += 512) {
        int m = idx >> 8, n = idx & 255;
        hsh[m * HSTR + n] = (t0 == 0) ? (unsigned short)0
                                      : h_state[(bg * 16 + m) * HID + n];
    }
    v8f creg;
#pragma unroll
    for (int v = 0; v < 8; ++v) {
        int m = v + laneHi * 8;
        creg[v] = (t0 == 0) ? 0.0f : c_state[(bg * 16 + m) * HID + col];
    }
    __syncthreads();

    for (int tl = 0; tl < TC; ++tl) {
        const int rblk = tl * 8 + bg;                // chunk row-block of this step
        v8f acc[4];
#pragma unroll
        for (int g = 0; g < 4; ++g) {
            const int cblk = (g << 4) + wave;
            const float4* s4 = reinterpret_cast<const float4*>(
                gx + ((size_t)rblk * NCB + cblk) * 256 + lane * 8);
            float4 a0 = s4[0], a1 = s4[1];
            acc[g][0] = a0.x; acc[g][1] = a0.y; acc[g][2] = a0.z; acc[g][3] = a0.w;
            acc[g][4] = a1.x; acc[g][5] = a1.y; acc[g][6] = a1.z; acc[g][7] = a1.w;
        }
        // Prefetch next step's tiles while the K-loop runs.
        if (tl + 1 < TC) {
#pragma unroll
            for (int g = 0; g < 4; ++g) {
                const float* nf = gx + ((size_t)(rblk + 8) * NCB + (g << 4) + wave) * 256
                                  + lane * 8;
                __builtin_prefetch(nf, 0, 0);
            }
        }
#pragma unroll
        for (int kb = 0; kb < HID / 32; ++kb) {
            const int k0 = kb * 32 + laneHi * 8;
            const uint4* ap = reinterpret_cast<const uint4*>(&hsh[lane15 * HSTR + k0]);
            ABFrag a;
            a.q[0] = ap[0];
            a.q[1] = ap[2];
#pragma unroll
            for (int g = 0; g < 4; ++g) {
                int n = (g << 8) + col;
                const uint4* wp =
                    reinterpret_cast<const uint4*>(Wt + (size_t)n * KTOT + DIM + k0);
                ABFrag b;
                b.q[0] = wp[0];
                b.q[1] = wp[2];
                acc[g] = __builtin_amdgcn_wmma_f32_16x16x32_bf16(
                    false, a.v, false, b.v, (short)0, acc[g], false, false);
            }
        }
        // Fused gate math (all registers).
        float hval[8];
#pragma unroll
        for (int v = 0; v < 8; ++v) {
            float fg = sigmoidf_(acc[0][v]);
            float ig = sigmoidf_(acc[1][v]);
            float ug = tanhf_(acc[2][v]);
            float og = sigmoidf_(acc[3][v]);
            float cv = fg * creg[v] + ig * ug;
            creg[v]  = cv;
            hval[v]  = og * tanhf_(cv);
        }
        __syncthreads();   // every wave finished reading old h
        const int t = t0 + tl;
#pragma unroll
        for (int v = 0; v < 8; ++v) {
            int m = v + laneHi * 8;
            hsh[m * HSTR + col] = f2bf(hval[v]);
            out[((size_t)t * BSZ + bg * 16 + m) * HID + col] = hval[v];
        }
        if (t == T_STEPS - 1) {
            float* hT = out + (size_t)T_STEPS * BSZ * HID;
            float* cT = hT + (size_t)BSZ * HID;
#pragma unroll
            for (int v = 0; v < 8; ++v) {
                int m = v + laneHi * 8;
                hT[(bg * 16 + m) * HID + col] = hval[v];
                cT[(bg * 16 + m) * HID + col] = creg[v];
            }
        }
        __syncthreads();   // new h fully visible before next step
    }

    // Persist state for next chunk.
#pragma unroll
    for (int v = 0; v < 8; ++v) {
        int m = v + laneHi * 8;
        c_state[(bg * 16 + m) * HID + col] = creg[v];
    }
    for (int idx = threadIdx.x; idx < 16 * HID; idx += 512) {
        int m = idx >> 8, n = idx & 255;
        h_state[(bg * 16 + m) * HID + n] = hsh[m * HSTR + n];
    }
}

// ------------------------------- launcher -----------------------------------
extern "C" void kernel_launch(void* const* d_in, const int* in_sizes, int n_in,
                              void* d_out, int out_size, void* d_ws, size_t ws_size,
                              hipStream_t stream) {
    (void)in_sizes; (void)n_in; (void)out_size; (void)ws_size;
    const float* x  = (const float*)d_in[0];
    const float* Wf = (const float*)d_in[1];
    const float* bf = (const float*)d_in[2];
    const float* Wi = (const float*)d_in[3];
    const float* bi = (const float*)d_in[4];
    const float* Wu = (const float*)d_in[5];
    const float* bu = (const float*)d_in[6];
    const float* Wo = (const float*)d_in[7];
    const float* bo = (const float*)d_in[8];
    float* out = (float*)d_out;

    char* ws = (char*)d_ws;
    unsigned short* Wt      = (unsigned short*)ws;                           // 1 MiB
    unsigned short* h_state = (unsigned short*)(ws + (1u << 20));            // 64 KiB
    float*          c_state = (float*)(ws + (1u << 20) + (1u << 16));        // 128 KiB
    float*          gx      = (float*)(ws + 1245184u);                       // 64 MiB
    unsigned short* xbf     = (unsigned short*)(ws + 1245184u + 67108864u);  // 64 MiB

    pack_weights<<<(NGATE * KTOT * HID + 255) / 256, 256, 0, stream>>>(Wf, Wi, Wu, Wo, Wt);
    {
        const size_t nelem = (size_t)T_STEPS * BSZ * DIM;           // 33.5M
        const int nthreads = (int)(nelem / 8);                      // 8 elems/thread
        xconvert<<<nthreads / 256, 256, 0, stream>>>(x, xbf);
    }

    for (int c = 0; c < NCHUNK; ++c) {
        const int t0 = c * TC;
        xproj_kernel<<<1024, 512, 0, stream>>>(xbf, Wt, bf, bi, bu, bo, gx, t0);
        lstm_scan<<<8, 512, 0, stream>>>(gx, Wt, out, h_state, c_state, t0);
    }
}
